// FlowmatchingActionHead_78073915507282
// MI455X (gfx1250) — matrix operand order repaired and checked by reading, not compiled, not run on hardware
//
#include <hip/hip_runtime.h>
#include <stdint.h>

// ---------------------------------------------------------------------------
// Types
// ---------------------------------------------------------------------------
typedef __bf16 v16bf __attribute__((ext_vector_type(16)));
typedef float  v8f   __attribute__((ext_vector_type(8)));
typedef unsigned int u32x4 __attribute__((ext_vector_type(4)));
typedef int          i32x4 __attribute__((ext_vector_type(4)));
typedef int          i32x8 __attribute__((ext_vector_type(8)));

static_assert(sizeof(v16bf) == 32, "v16bf must be 32 bytes");

#define NUM_CATS 32
#define HIDDEN   1024
#define N_TOKENS 2048

// ---------------------------------------------------------------------------
// TDM: issue a 32(K) x 128(N) fp32 tile load, row stride 1024 floats,
// into LDS at ldsdst. Descriptor per CDNA5 ISA ch.8 (D# groups 0/1).
// ---------------------------------------------------------------------------
__device__ __forceinline__ void tdm_issue_tile(const float* gsrc, const void* ldsdst, int Kdim) {
  unsigned long long ga = (unsigned long long)(uintptr_t)gsrc;
  unsigned lds = (unsigned)(uintptr_t)ldsdst;   // low 32 bits of generic ptr = LDS offset
  u32x4 g0;
  g0[0] = 1u;                                        // count=1, user descriptor
  g0[1] = lds;                                       // lds_addr [63:32]
  g0[2] = (unsigned)ga;                              // global_addr [95:64]
  g0[3] = (unsigned)((ga >> 32) & 0x01FFFFFFu)       // global_addr [120:96]
        | (2u << 30);                                // type=2 ("image")
  i32x8 g1;
  g1[0] = (2 << 16);                                 // data_size=4B; no multicast/pad
  g1[1] = (int)(1024u << 16);                        // tensor_dim0 lo16 = 1024 (N)
  g1[2] = (int)(((unsigned)Kdim & 0xFFFFu) << 16);   // tensor_dim0 hi16=0 | tensor_dim1 lo16 = K
  g1[3] = (int)((((unsigned)Kdim >> 16) & 0xFFFFu)   // tensor_dim1 hi16
        | (128u << 16));                             // tile_dim0 = 128
  g1[4] = 32;                                        // tile_dim1 = 32, tile_dim2 = 0
  g1[5] = 1024;                                      // tensor_dim0_stride lo32 = 1024
  g1[6] = 0;                                         // stride0 hi16 | stride1 lo16
  g1[7] = 0;                                         // stride1 hi32
  i32x4 z4 = {};
#if defined(__clang_major__) && (__clang_major__ >= 23)
  i32x8 z8 = {};
  __builtin_amdgcn_tensor_load_to_lds(g0, g1, z4, z4, z8, 0);
#else
  __builtin_amdgcn_tensor_load_to_lds(g0, g1, z4, z4, 0);
#endif
}

// Split fp32 -> (hi, lo) bf16 pair:  x ~= hi + lo, |lo| <= ulp(hi)/2
__device__ __forceinline__ void bf16_split(float x, __bf16& h, __bf16& l) {
  h = (__bf16)x;
  l = (__bf16)(x - (float)h);
}

// ---------------------------------------------------------------------------
// Kernel 1: build permutation (tokens grouped by category) + offsets
// ---------------------------------------------------------------------------
__global__ __launch_bounds__(256) void fm_prep(const int* __restrict__ cat_ids,
                                               int* __restrict__ perm,
                                               int* __restrict__ cat_off) {
  __shared__ int cnt[NUM_CATS];
  __shared__ int base[NUM_CATS + 1];
  const int tid = threadIdx.x;
  if (tid < NUM_CATS) cnt[tid] = 0;
  __syncthreads();
  for (int n = tid; n < N_TOKENS; n += 256) atomicAdd(&cnt[cat_ids[n]], 1);
  __syncthreads();
  if (tid == 0) {
    int s = 0;
    for (int c = 0; c < NUM_CATS; ++c) { base[c] = s; s += cnt[c]; }
    base[NUM_CATS] = s;
  }
  __syncthreads();
  if (tid < NUM_CATS + 1) cat_off[tid] = base[tid];
  if (tid < NUM_CATS) cnt[tid] = base[tid];
  __syncthreads();
  for (int n = tid; n < N_TOKENS; n += 256) {
    int slot = atomicAdd(&cnt[cat_ids[n]], 1);
    perm[slot] = n;
  }
}

// ---------------------------------------------------------------------------
// Kernel 2: permuted fp32 actions (ap) + sinusoidal tau into xp[:,1024:]
// ---------------------------------------------------------------------------
__global__ __launch_bounds__(256) void fm_pack(const float* __restrict__ actions,
                                               const float* __restrict__ timesteps,
                                               const int* __restrict__ perm,
                                               float* __restrict__ ap,
                                               float* __restrict__ xp) {
  const int i = blockIdx.x;      // permuted row
  const int n = perm[i];         // original token
  const float t = timesteps[n & 127];
  const int tid = threadIdx.x;
  if (tid < 32) ap[i * 32 + tid] = actions[n * 32 + tid];
  const float kf = -9.210340371976184f / 512.0f;   // -ln(10000)/512
  for (int d = tid; d < HIDDEN; d += 256) {
    const int j = (d < 512) ? d : d - 512;
    const float f = __expf(kf * (float)j);
    const float arg = t * f;
    const float v = (d < 512) ? __sinf(arg) : __cosf(arg);
    xp[(size_t)i * 2048 + HIDDEN + d] = v;
  }
}

// ---------------------------------------------------------------------------
// Kernel 3: category-routed GEMM with split-bf16 (near-fp32) accumulation.
//   Block = 256 thr = 8 waves, tile M=64 x N=128, K-step 32.
//   Per 16x16 subtile & K-step: 3 WMMAs (ah*bh + al*bh + ah*bl), fp32 accum.
//   MODE 0: out fp32, stride 2048  (a_emb -> xp[:, :1024])
//   MODE 1: swish, out fp32, stride 1024 (h -> hp)
//   MODE 2: out fp32, scatter rows via perm (final output)
// ---------------------------------------------------------------------------
template <int K, int DIN, int MODE>
__global__ __launch_bounds__(256)
void fm_gemm(const float* __restrict__ Xp, const float* __restrict__ W,
             const float* __restrict__ Bias, float* __restrict__ Out,
             const int* __restrict__ cat_off, const int* __restrict__ perm) {
  const int c  = blockIdx.y;
  const int n0 = blockIdx.x * 128;
  const int rbeg = cat_off[c];
  const int rend = cat_off[c + 1];
  const int m0 = rbeg + (int)blockIdx.z * 64;
  if (m0 >= rend) return;
  const int mrows = (rend - m0 < 64) ? (rend - m0) : 64;

  __shared__ __attribute__((aligned(16))) float  wbuf[2][32 * 128]; // fp32 weights (TDM dest) 32KB
  __shared__ __attribute__((aligned(16))) __bf16 abuf_hi[64 * 40];  // A hi, padded stride 40
  __shared__ __attribute__((aligned(16))) __bf16 abuf_lo[64 * 40];  // A lo
  __shared__ __attribute__((aligned(16))) __bf16 bpk_hi[128 * 40];  // B hi, [n][k] padded
  __shared__ __attribute__((aligned(16))) __bf16 bpk_lo[128 * 40];  // B lo

  const int tid  = threadIdx.x;
  const int wave = tid >> 5;
  const int lane = tid & 31;
  const int hi   = lane >> 4;
  const int nl   = lane & 15;
  const int mg   = wave & 3;
  const int ng   = wave >> 2;

  v8f acc[4] = {v8f{}, v8f{}, v8f{}, v8f{}};

  const float* wsrc = W + (size_t)c * K * HIDDEN + n0;
  if (wave == 0) tdm_issue_tile(wsrc, &wbuf[0][0], K);

  constexpr int KT = K / 32;
  for (int kt = 0; kt < KT; ++kt) {
    const int k = kt * 32;
    if (wave == 0 && kt + 1 < KT)
      tdm_issue_tile(wsrc + (size_t)(k + 32) * HIDDEN, &wbuf[(kt + 1) & 1][0], K);

    // Stage A: 64 x 32 fp32 activations -> hi/lo bf16 LDS tiles
    {
      const int r   = tid >> 2;
      const int seg = (tid & 3) * 8;                  // 8 floats per thread
      const int rr  = (r < mrows) ? r : (mrows - 1);
      const float* src = Xp + (size_t)(m0 + rr) * DIN + k + seg;
      float f[8];
      *(float4*)&f[0] = *(const float4*)(src);
      *(float4*)&f[4] = *(const float4*)(src + 4);
      __bf16 fh[8], fl[8];
#pragma unroll
      for (int q = 0; q < 8; ++q) bf16_split(f[q], fh[q], fl[q]);
      *(uint4*)&abuf_hi[r * 40 + seg] = *(const uint4*)&fh[0];
      *(uint4*)&abuf_lo[r * 40 + seg] = *(const uint4*)&fl[0];
    }
    if (wave == 0) {
      if (kt + 1 < KT) __builtin_amdgcn_s_wait_tensorcnt(1);  // oldest TDM (this tile) done
      else             __builtin_amdgcn_s_wait_tensorcnt(0);
    }
    __syncthreads();

    // Stage pack: fp32 weights -> hi/lo bf16, transposed to [n][k]
    {
      const float* wb = &wbuf[kt & 1][0];
      const int n  = tid >> 1;
      const int kh = (tid & 1) * 16;
      __bf16 thi[16], tlo[16];
#pragma unroll
      for (int j2 = 0; j2 < 16; ++j2) bf16_split(wb[(kh + j2) * 128 + n], thi[j2], tlo[j2]);
      *(uint4*)&bpk_hi[n * 40 + kh]     = *(const uint4*)&thi[0];
      *(uint4*)&bpk_hi[n * 40 + kh + 8] = *(const uint4*)&thi[8];
      *(uint4*)&bpk_lo[n * 40 + kh]     = *(const uint4*)&tlo[0];
      *(uint4*)&bpk_lo[n * 40 + kh + 8] = *(const uint4*)&tlo[8];
    }
    __syncthreads();

    // Stage compute: fragments in this kernel's contiguous k order (k = 16*hi + e), same for A and B
    {
      union FR { uint4 q[2]; v16bf v; };
      const int m = mg * 16 + nl;
      // A frag lane(L): row m=L%16; elem e -> k = e + 16*hi  (contiguous in abuf)
      FR ah, al;
      ah.q[0] = *(const uint4*)&abuf_hi[m * 40 + 16 * hi];
      ah.q[1] = *(const uint4*)&abuf_hi[m * 40 + 16 * hi + 8];
      al.q[0] = *(const uint4*)&abuf_lo[m * 40 + 16 * hi];
      al.q[1] = *(const uint4*)&abuf_lo[m * 40 + 16 * hi + 8];
#pragma unroll
      for (int j = 0; j < 4; ++j) {
        const int nsub = ng * 64 + j * 16 + nl;
        // B frag lane(L): col n=L%16; elem e -> k = e + 16*hi  (contiguous in bpk)
        FR bh, bl;
        bh.q[0] = *(const uint4*)&bpk_hi[nsub * 40 + 16 * hi];
        bh.q[1] = *(const uint4*)&bpk_hi[nsub * 40 + 16 * hi + 8];
        bl.q[0] = *(const uint4*)&bpk_lo[nsub * 40 + 16 * hi];
        bl.q[1] = *(const uint4*)&bpk_lo[nsub * 40 + 16 * hi + 8];
        acc[j] = __builtin_amdgcn_wmma_f32_16x16x32_bf16(false, al.v, false, bh.v,
                                                         (short)0, acc[j], false, false);
        acc[j] = __builtin_amdgcn_wmma_f32_16x16x32_bf16(false, ah.v, false, bl.v,
                                                         (short)0, acc[j], false, false);
        acc[j] = __builtin_amdgcn_wmma_f32_16x16x32_bf16(false, ah.v, false, bh.v,
                                                         (short)0, acc[j], false, false);
      }
    }
    __syncthreads();
  }

  // Epilogue. C/D layout: lane L -> col n=L%16, VGPR v -> row v + 8*(L/16)
  if (mrows == 64) {
    // fast path: full tile, no per-row predication
#pragma unroll
    for (int j = 0; j < 4; ++j) {
      const int nsub = ng * 64 + j * 16 + nl;
      const float bv = Bias[c * HIDDEN + n0 + nsub];
#pragma unroll
      for (int v = 0; v < 8; ++v) {
        const int row = m0 + mg * 16 + v + 8 * hi;
        float val = acc[j][v] + bv;
        if (MODE == 1) val = val * (1.0f / (1.0f + __expf(-val)));  // swish
        if (MODE == 2) {
          Out[(size_t)perm[row] * HIDDEN + n0 + nsub] = val;
        } else {
          const size_t stride = (MODE == 0) ? 2048 : 1024;
          Out[(size_t)row * stride + n0 + nsub] = val;
        }
      }
    }
  } else {
#pragma unroll
    for (int j = 0; j < 4; ++j) {
      const int nsub = ng * 64 + j * 16 + nl;
      const float bv = Bias[c * HIDDEN + n0 + nsub];
#pragma unroll
      for (int v = 0; v < 8; ++v) {
        const int mrel = mg * 16 + v + 8 * hi;
        if (mrel >= mrows) continue;
        const int row = m0 + mrel;
        float val = acc[j][v] + bv;
        if (MODE == 1) val = val * (1.0f / (1.0f + __expf(-val)));
        if (MODE == 2) {
          Out[(size_t)perm[row] * HIDDEN + n0 + nsub] = val;
        } else {
          const size_t stride = (MODE == 0) ? 2048 : 1024;
          Out[(size_t)row * stride + n0 + nsub] = val;
        }
      }
    }
  }
}

// ---------------------------------------------------------------------------
// Launcher
// ---------------------------------------------------------------------------
extern "C" void kernel_launch(void* const* d_in, const int* in_sizes, int n_in,
                              void* d_out, int out_size, void* d_ws, size_t ws_size,
                              hipStream_t stream) {
  (void)in_sizes; (void)n_in; (void)out_size; (void)ws_size;
  const float* actions   = (const float*)d_in[0];
  const float* timesteps = (const float*)d_in[1];
  const int*   cat_ids   = (const int*)d_in[2];
  const float* W1 = (const float*)d_in[3];
  const float* b1 = (const float*)d_in[4];
  const float* W2 = (const float*)d_in[5];
  const float* b2 = (const float*)d_in[6];
  const float* W3 = (const float*)d_in[7];
  const float* b3 = (const float*)d_in[8];
  float* out = (float*)d_out;

  char* ws = (char*)d_ws;
  int*   perm    = (int*)ws;                                  // 2048 ints
  int*   cat_off = (int*)(ws + 8192);                         // 33 ints
  float* ap      = (float*)(ws + 16384);                      // 2048*32 f32   (256 KB)
  float* xp      = (float*)(ws + 16384 + 262144);             // 2048*2048 f32 (16 MB)
  float* hp      = (float*)(ws + 16384 + 262144 + (size_t)16 * 1024 * 1024); // 8 MB

  fm_prep<<<1, 256, 0, stream>>>(cat_ids, perm, cat_off);
  fm_pack<<<N_TOKENS, 256, 0, stream>>>(actions, timesteps, perm, ap, xp);

  dim3 grid(HIDDEN / 128, NUM_CATS, N_TOKENS / 64);
  dim3 blk(256);
  // GEMM1: a_emb = ap @ W1 + b1 -> xp[:, :1024]
  fm_gemm<32, 32, 0><<<grid, blk, 0, stream>>>(ap, W1, b1, xp, cat_off, perm);
  // GEMM2: h = swish(xp @ W2 + b2) -> hp
  fm_gemm<2048, 2048, 1><<<grid, blk, 0, stream>>>(xp, W2, b2, hp, cat_off, perm);
  // GEMM3: out[perm[i]] = hp @ W3 + b3
  fm_gemm<1024, 1024, 2><<<grid, blk, 0, stream>>>(hp, W3, b3, out, cat_off, perm);
}
